// PointNetSetAbstraction_3143916061381
// MI455X (gfx1250) — matrix operand order repaired and checked by reading, not compiled, not run on hardware
//
#include <hip/hip_runtime.h>

typedef _Float16 half_t;
typedef __attribute__((ext_vector_type(16))) _Float16 v16h;
typedef __attribute__((ext_vector_type(8)))  _Float16 v8h;
typedef __attribute__((ext_vector_type(4)))  _Float16 v4h;
typedef __attribute__((ext_vector_type(8)))  float    v8f;
typedef __attribute__((ext_vector_type(4)))  float    v4f;

#define BATCH    8
#define NPTS     8192
#define NPOINT_  1024
#define NSAMPLE_ 32
#define IN_CH_   64
#define RADIUS2  0.04f
#define EPS_     1e-5f
#define TOKENS   (BATCH * NPOINT_ * NSAMPLE_)   /* 262144 */
#define K0PAD    96                              /* 67 padded to 3*32 */
#define STAT_BLOCKS 256

// ---------------------------------------------------------------------------
// 1) Farthest point sampling: one block per batch; per-thread distances live
//    in registers; wave32 shuffle argmax + one 8-entry cross-wave reduce.
//    Only 2 __syncthreads per sequential step.
// ---------------------------------------------------------------------------
__global__ void fps_kernel(const float* __restrict__ xyz, float* __restrict__ new_xyz) {
  const int b    = blockIdx.x;
  const int tid  = threadIdx.x;                 // 256 threads = 8 waves
  const int lane = tid & 31;
  const int wid  = tid >> 5;
  const float* x = xyz + (size_t)b * NPTS * 3;

  float dist[NPTS / 256];
#pragma unroll
  for (int j = 0; j < NPTS / 256; ++j) dist[j] = 1e10f;

  __shared__ float cx, cy, cz;
  __shared__ float swval[8];
  __shared__ int   swidx[8];
  __shared__ int   sfar;
  if (tid == 0) sfar = 0;
  __syncthreads();

  for (int it = 0; it < NPOINT_; ++it) {
    if (tid == 0) {
      const int f = sfar;
      const float a0 = x[f * 3 + 0], a1 = x[f * 3 + 1], a2 = x[f * 3 + 2];
      cx = a0; cy = a1; cz = a2;
      float* o = new_xyz + ((size_t)b * NPOINT_ + it) * 3;
      o[0] = a0; o[1] = a1; o[2] = a2;
    }
    __syncthreads();

    float lmax = -1.0f; int lidx = 0;
#pragma unroll
    for (int j = 0; j < NPTS / 256; ++j) {
      const int p = j * 256 + tid;
      const float dx = x[p * 3 + 0] - cx;
      const float dy = x[p * 3 + 1] - cy;
      const float dz = x[p * 3 + 2] - cz;
      const float d  = dx * dx + dy * dy + dz * dz;
      const float nd = fminf(dist[j], d);
      dist[j] = nd;
      if (nd > lmax) { lmax = nd; lidx = p; }
    }
    // wave32 argmax reduction (ds_swizzle / permlane path)
#pragma unroll
    for (int off = 16; off > 0; off >>= 1) {
      const float ov = __shfl_down(lmax, off, 32);
      const int   oi = __shfl_down(lidx, off, 32);
      if (ov > lmax) { lmax = ov; lidx = oi; }
    }
    if (lane == 0) { swval[wid] = lmax; swidx[wid] = lidx; }
    __syncthreads();
    if (tid == 0) {
      float bv = swval[0]; int bi = swidx[0];
#pragma unroll
      for (int w = 1; w < 8; ++w)
        if (swval[w] > bv) { bv = swval[w]; bi = swidx[w]; }
      sfar = bi;
    }
    // no extra barrier needed: next iteration's first barrier orders sfar/swval
  }
}

// ---------------------------------------------------------------------------
// 2) Ball query + gather + concat -> padded f16 A0[TOKENS][96].
//    One wave32 per centroid: ballot + prefix-popc keeps the first 32
//    in-radius indices in ascending point order. Row stores vectorized b128.
// ---------------------------------------------------------------------------
__global__ void group_kernel(const float* __restrict__ xyz, const float* __restrict__ pts,
                             const float* __restrict__ new_xyz, half_t* __restrict__ A) {
  const int wave = threadIdx.x >> 5;           // 4 waves / block
  const int lane = threadIdx.x & 31;
  const int g = blockIdx.x * 4 + wave;         // 0 .. BATCH*NPOINT-1
  const int b = g / NPOINT_;
  const float* x = xyz + (size_t)b * NPTS * 3;
  const float* f = pts + (size_t)b * NPTS * IN_CH_;
  const float* c = new_xyz + (size_t)g * 3;
  const float cx = c[0], cy = c[1], cz = c[2];

  __shared__ int sidx[4][NSAMPLE_];

  int cnt = 0;
  for (int base = 0; base < NPTS && cnt < NSAMPLE_; base += 32) {
    const int p = base + lane;
    const float dx = x[p * 3 + 0] - cx;
    const float dy = x[p * 3 + 1] - cy;
    const float dz = x[p * 3 + 2] - cz;
    const bool in = (dx * dx + dy * dy + dz * dz) <= RADIUS2;
    const unsigned mask = __builtin_amdgcn_ballot_w32(in);
    if (in) {
      const int pos = cnt + __popc(mask & ((1u << lane) - 1u));
      if (pos < NSAMPLE_) sidx[wave][pos] = p;
    }
    cnt += __popc(mask);
  }
  if (cnt > NSAMPLE_) cnt = NSAMPLE_;
  __syncthreads();   // all waves converge here exactly once

  const int first = sidx[wave][0];             // cnt >= 1 (centroid is a point)
  const int idx = (lane < cnt) ? sidx[wave][lane] : first;

  half_t buf[K0PAD];
  buf[0] = (half_t)(x[idx * 3 + 0] - cx);
  buf[1] = (half_t)(x[idx * 3 + 1] - cy);
  buf[2] = (half_t)(x[idx * 3 + 2] - cz);
  const float* fp = f + (size_t)idx * IN_CH_;
#pragma unroll
  for (int j = 0; j < IN_CH_; ++j) buf[3 + j] = (half_t)fp[j];
#pragma unroll
  for (int j = 3 + IN_CH_; j < K0PAD; ++j) buf[j] = (half_t)0.0f;

  half_t* out = A + (size_t)(g * NSAMPLE_ + lane) * K0PAD;   // 192B row, 16B aligned
#pragma unroll
  for (int j = 0; j < K0PAD; j += 8)
    *(v8h*)(out + j) = *(const v8h*)(buf + j);
}

// ---------------------------------------------------------------------------
// 3) Pack f32 weights [cout][cin] into zero-padded f16 [cout][kpad].
// ---------------------------------------------------------------------------
__global__ void pack_w_kernel(const float* __restrict__ W, half_t* __restrict__ Wh,
                              int cout, int cin, int kpad) {
  const int i = blockIdx.x * blockDim.x + threadIdx.x;
  if (i >= cout * kpad) return;
  const int o = i / kpad, k = i % kpad;
  Wh[i] = (k < cin) ? (half_t)W[o * cin + k] : (half_t)0.0f;
}

// ---------------------------------------------------------------------------
// 4) WMMA GEMM: C[T][COUT] = A[T][K](f16) * Wh[COUT][K]^T + bias, f32 accum.
//    One wave computes a 16 x COUT row-strip: the A fragment is loaded once
//    per K-chunk and reused across all COUT/16 N-tiles (register reuse).
//    Per-lane addressing follows the ISA 16-bit A / B / f32 D VGPR layouts.
// ---------------------------------------------------------------------------
template <int K, int COUT>
__global__ void gemm_wmma_kernel(const half_t* __restrict__ A, const half_t* __restrict__ Wh,
                                 const float* __restrict__ bias, float* __restrict__ C) {
  constexpr int NT = COUT / 16;                 // N tiles per wave
  const int lane = threadIdx.x & 31;
  const int tM   = (int)((blockIdx.x * blockDim.x + threadIdx.x) >> 5); // one M-tile per wave
  const int l15  = lane & 15;
  const int hi   = lane >> 4;                   // lane half: 0 or 1

  const half_t* ap = A + (size_t)(tM * 16 + l15) * K + hi * 8;

  union V16 { v16h v; v8h h[2]; };
  v8f acc[NT];
#pragma unroll
  for (int t = 0; t < NT; ++t) acc[t] = (v8f){};

#pragma unroll
  for (int k0 = 0; k0 < K; k0 += 32) {
    __builtin_prefetch(ap + k0 + 64, 0, 3);     // global_prefetch_b8 (near-temporal)
    V16 a;
    a.h[0] = *(const v8h*)(ap + k0);            // K = k0 + hi*8 + 0..7
    a.h[1] = *(const v8h*)(ap + k0 + 16);       // K = k0 + hi*8 + 16..23
#pragma unroll
    for (int t = 0; t < NT; ++t) {
      const half_t* wp = Wh + (size_t)(t * 16 + l15) * K + hi * 8;
      V16 w;
      w.h[0] = *(const v8h*)(wp + k0);          // K = k0 + hi*8 + 0..7
      w.h[1] = *(const v8h*)(wp + k0 + 16);     // K = k0 + hi*8 + 16..23
      acc[t] = __builtin_amdgcn_wmma_f32_16x16x32_f16(
          /*neg_a=*/false, a.v, /*neg_b=*/false, w.v,
          /*c_mod=*/(short)0, acc[t], /*reuse_a=*/false, /*reuse_b=*/false);
    }
  }

  const int mbase = tM * 16 + hi * 8;
#pragma unroll
  for (int t = 0; t < NT; ++t) {
    const int col  = t * 16 + l15;
    const float bv = bias[col];
#pragma unroll
    for (int r = 0; r < 8; ++r)
      C[(size_t)(mbase + r) * COUT + col] = acc[t][r] + bv;
  }
}

// ---------------------------------------------------------------------------
// 5a) Per-channel partial sums over a contiguous slab of rows (coalesced).
//     Deterministic two-pass reduction (no float atomics).
// ---------------------------------------------------------------------------
__global__ void colstats_partial_kernel(const float* __restrict__ X,
                                        float* __restrict__ partial, int Cout) {
  const int tid = threadIdx.x;                  // 256 threads
  const int blk = blockIdx.x;                   // STAT_BLOCKS blocks
  const int tpc = 256 / Cout;                   // threads per column (4 or 2)
  const int col = tid % Cout;
  const int sub = tid / Cout;
  const int rows = TOKENS / STAT_BLOCKS;        // 1024 rows per block
  const int r0 = blk * rows;

  float s = 0.0f, s2 = 0.0f;
  for (int r = sub; r < rows; r += tpc) {
    const float v = X[(size_t)(r0 + r) * Cout + col];
    s += v; s2 += v * v;
  }
  __shared__ float sh1[256], sh2[256];
  sh1[tid] = s; sh2[tid] = s2;
  __syncthreads();
  if (sub == 0) {
    for (int j = 1; j < tpc; ++j) { s += sh1[col + j * Cout]; s2 += sh2[col + j * Cout]; }
    partial[(size_t)blk * 2 * Cout + col]        = s;
    partial[(size_t)blk * 2 * Cout + Cout + col] = s2;
  }
}

// ---------------------------------------------------------------------------
// 5b) Finalize: reduce partials -> mean / inv-std per channel.
// ---------------------------------------------------------------------------
__global__ void colstats_final_kernel(const float* __restrict__ partial,
                                      float* __restrict__ stats, int Cout) {
  const int c = blockIdx.x * blockDim.x + threadIdx.x;
  if (c >= Cout) return;
  float s = 0.0f, s2 = 0.0f;
  for (int b = 0; b < STAT_BLOCKS; ++b) {
    s  += partial[(size_t)b * 2 * Cout + c];
    s2 += partial[(size_t)b * 2 * Cout + Cout + c];
  }
  const float mean = s / (float)TOKENS;
  const float var  = s2 / (float)TOKENS - mean * mean;
  stats[2 * c]     = mean;
  stats[2 * c + 1] = rsqrtf(var + EPS_);
}

// ---------------------------------------------------------------------------
// 6) BN + ReLU, repack to f16 activations for the next WMMA layer (4-wide).
// ---------------------------------------------------------------------------
__global__ void bn_relu_f16_kernel(const float* __restrict__ X, const float* __restrict__ stats,
                                   const float* __restrict__ g, const float* __restrict__ beta,
                                   half_t* __restrict__ A, int Cout) {
  const size_t base = ((size_t)blockIdx.x * blockDim.x + threadIdx.x) * 4;
  const int c0 = (int)(base % (size_t)Cout);    // Cout % 4 == 0 -> same row
  const v4f xv = *(const v4f*)(X + base);
  v4h o;
#pragma unroll
  for (int j = 0; j < 4; ++j) {
    const int c = c0 + j;
    float v = (xv[j] - stats[2 * c]) * stats[2 * c + 1] * g[c] + beta[c];
    o[j] = (half_t)fmaxf(v, 0.0f);
  }
  *(v4h*)(A + base) = o;
}

// ---------------------------------------------------------------------------
// 7) Final layer epilogue: BN + ReLU + max over NSAMPLE -> new_points.
// ---------------------------------------------------------------------------
__global__ void bn_relu_max_kernel(const float* __restrict__ X, const float* __restrict__ stats,
                                   const float* __restrict__ g, const float* __restrict__ beta,
                                   float* __restrict__ out, int Cout) {
  const int i = blockIdx.x * blockDim.x + threadIdx.x;  // over B*NPOINT*Cout
  const int c   = i % Cout;
  const int grp = i / Cout;
  const float mean = stats[2 * c], rinv = stats[2 * c + 1];
  const float gg = g[c], bb = beta[c];
  const float* p = X + (size_t)grp * NSAMPLE_ * Cout + c;
  float m = 0.0f;                                       // relu output >= 0
#pragma unroll
  for (int k = 0; k < NSAMPLE_; ++k) {
    const float v = (p[(size_t)k * Cout] - mean) * rinv * gg + bb;
    m = fmaxf(m, fmaxf(v, 0.0f));
  }
  out[i] = m;
}

// ---------------------------------------------------------------------------
extern "C" void kernel_launch(void* const* d_in, const int* in_sizes, int n_in,
                              void* d_out, int out_size, void* d_ws, size_t ws_size,
                              hipStream_t stream) {
  const float* xyz    = (const float*)d_in[0];
  const float* points = (const float*)d_in[1];
  const float* W0  = (const float*)d_in[2];
  const float* b0  = (const float*)d_in[3];
  const float* g0  = (const float*)d_in[4];
  const float* be0 = (const float*)d_in[5];
  const float* W1  = (const float*)d_in[6];
  const float* b1  = (const float*)d_in[7];
  const float* g1  = (const float*)d_in[8];
  const float* be1 = (const float*)d_in[9];
  const float* W2  = (const float*)d_in[10];
  const float* b2  = (const float*)d_in[11];
  const float* g2  = (const float*)d_in[12];
  const float* be2 = (const float*)d_in[13];

  float* out        = (float*)d_out;
  float* new_xyz    = out;                               // 8*1024*3
  float* new_points = out + (size_t)BATCH * NPOINT_ * 3; // 8*1024*128

  char* ws = (char*)d_ws;
  size_t off = 0;
  auto align256 = [](size_t v) { return (v + 255) & ~(size_t)255; };

  half_t* A = (half_t*)(ws + off);
  off = align256(off + (size_t)TOKENS * K0PAD * sizeof(half_t));       // ~50 MB
  float* G = (float*)(ws + off);
  off = align256(off + (size_t)TOKENS * 128 * sizeof(float));          // ~134 MB
  float* stats = (float*)(ws + off);
  off = align256(off + 256 * sizeof(float));
  float* partial = (float*)(ws + off);
  off = align256(off + (size_t)STAT_BLOCKS * 2 * 128 * sizeof(float)); // 256 KB
  half_t* Wh0 = (half_t*)(ws + off); off = align256(off + 64 * 96 * sizeof(half_t));
  half_t* Wh1 = (half_t*)(ws + off); off = align256(off + 64 * 64 * sizeof(half_t));
  half_t* Wh2 = (half_t*)(ws + off); off = align256(off + 128 * 64 * sizeof(half_t));

  // 1) FPS -> new_xyz (output + consumed by grouping)
  fps_kernel<<<BATCH, 256, 0, stream>>>(xyz, new_xyz);

  // 2) Ball query + gather + concat -> A[TOKENS][96] f16
  group_kernel<<<BATCH * NPOINT_ / 4, 128, 0, stream>>>(xyz, points, new_xyz, A);

  // 3) f16 weight packs
  pack_w_kernel<<<(64 * 96 + 255) / 256, 256, 0, stream>>>(W0, Wh0, 64, 67, 96);
  pack_w_kernel<<<(64 * 64 + 255) / 256, 256, 0, stream>>>(W1, Wh1, 64, 64, 64);
  pack_w_kernel<<<(128 * 64 + 255) / 256, 256, 0, stream>>>(W2, Wh2, 128, 64, 64);

  constexpr int GEMM_BLOCKS = (TOKENS / 16) / 8;         // 8 waves/block, 1 M-tile/wave

  // Layer 0: 96 -> 64
  gemm_wmma_kernel<96, 64><<<GEMM_BLOCKS, 256, 0, stream>>>(A, Wh0, b0, G);
  colstats_partial_kernel<<<STAT_BLOCKS, 256, 0, stream>>>(G, partial, 64);
  colstats_final_kernel<<<1, 128, 0, stream>>>(partial, stats, 64);
  bn_relu_f16_kernel<<<TOKENS * 64 / 4 / 256, 256, 0, stream>>>(G, stats, g0, be0, A, 64);

  // Layer 1: 64 -> 64
  gemm_wmma_kernel<64, 64><<<GEMM_BLOCKS, 256, 0, stream>>>(A, Wh1, b1, G);
  colstats_partial_kernel<<<STAT_BLOCKS, 256, 0, stream>>>(G, partial, 64);
  colstats_final_kernel<<<1, 128, 0, stream>>>(partial, stats, 64);
  bn_relu_f16_kernel<<<TOKENS * 64 / 4 / 256, 256, 0, stream>>>(G, stats, g1, be1, A, 64);

  // Layer 2: 64 -> 128, then BN+ReLU+max fused into the output write
  gemm_wmma_kernel<64, 128><<<GEMM_BLOCKS, 256, 0, stream>>>(A, Wh2, b2, G);
  colstats_partial_kernel<<<STAT_BLOCKS, 256, 0, stream>>>(G, partial, 128);
  colstats_final_kernel<<<1, 128, 0, stream>>>(partial, stats, 128);
  bn_relu_max_kernel<<<BATCH * NPOINT_ * 128 / 256, 256, 0, stream>>>(G, stats, g2, be2,
                                                                      new_points, 128);
}
